// Attention_75539884802496
// MI455X (gfx1250) — compile-verified, matrix operand order checked by reading, not run on hardware
//
#include <hip/hip_runtime.h>
#include <hip/hip_bf16.h>

// ---------------------------------------------------------------------------
// Problem constants (from reference setup_inputs)
// ---------------------------------------------------------------------------
#define B_    1
#define S_    56320          // 16*44*80
#define H_    8
#define D_    64
#define P_    128
#define R_    2
#define NSEG  3              // global + 2 regions
#define NKEY  384            // NSEG * P_
#define T_    121
#define MH_   88
#define MW_   160
#define SCALE 0.125f         // 64^-0.5 (exact power of two -> foldable into Q)
#define NKT   24             // NKEY/16 key tiles
#define QROWS 16             // rows per wave
#define WAVES 4              // waves per block
#define QBLK  (QROWS*WAVES)  // 64 queries per block

typedef __attribute__((ext_vector_type(16))) __bf16 v16bf;
typedef __attribute__((ext_vector_type(8)))  float  v8f;

union Frag { v16bf v; unsigned u[8]; };

// Round-to-nearest-even f32 -> bf16 pair pack (used in one-time pack kernels).
static __device__ inline unsigned pack_bf16_rne(float a, float b) {
    union { float f; unsigned u; } va, vb; va.f = a; vb.f = b;
    unsigned ua = va.u + (0x7FFFu + ((va.u >> 16) & 1u));
    unsigned ub = vb.u + (0x7FFFu + ((vb.u >> 16) & 1u));
    return (ua >> 16) | (ub & 0xFFFF0000u);
}
static __device__ inline Frag load_frag(const unsigned* __restrict__ p) {
    Frag f;
    f.v = *(const v16bf*)p;   // 32B -> two global_load_b128 straight into operand regs
    return f;
}
static __device__ inline v8f wmma_bf16(v16bf a, v16bf b, v8f c) {
    return __builtin_amdgcn_wmma_f32_16x16x32_bf16(
        /*neg_a=*/false, a, /*neg_b=*/false, b,
        /*c_mod=*/(short)0, c, /*reuse_a=*/false, /*reuse_b=*/false);
}

// ---------------------------------------------------------------------------
// Kernel 1: trilinear-downsample region masks -> per-query allowed-segment bits
//   bit0 = no region covers (global segment allowed)
//   bit(r+1) = region r covers this latent position
// Exact align_corners=False weights: spatial 2x (avg of 2x2),
// temporal: t==0 -> frame 0 only; t>=1 -> frames (8t-4, 8t-3) at 0.5/0.5.
// ---------------------------------------------------------------------------
__global__ void mask_kernel(const float* __restrict__ rm, unsigned* __restrict__ allowed) {
    int s = blockIdx.x * blockDim.x + threadIdx.x;
    if (s >= S_) return;
    int t   = s / (44 * 80);
    int rem = s % (44 * 80);
    int y = rem / 80, x = rem % 80;
    int yy = 2 * y, xx = 2 * x;
    unsigned bits = 0;
    bool any = false;
    #pragma unroll
    for (int r = 0; r < R_; ++r) {
        const float* base = rm + (size_t)r * T_ * MH_ * MW_;
        float val;
        if (t == 0) {
            const float* f = base;  // frame 0, weight 1
            val = 0.25f * (f[yy*MW_+xx] + f[yy*MW_+xx+1] +
                           f[(yy+1)*MW_+xx] + f[(yy+1)*MW_+xx+1]);
        } else {
            int f0 = 1 + 8 * (t - 1) + 3;
            const float* fa = base + (size_t)f0 * MH_ * MW_;
            const float* fb = fa + MH_ * MW_;
            val = 0.125f * (fa[yy*MW_+xx] + fa[yy*MW_+xx+1] +
                            fa[(yy+1)*MW_+xx] + fa[(yy+1)*MW_+xx+1] +
                            fb[yy*MW_+xx] + fb[yy*MW_+xx+1] +
                            fb[(yy+1)*MW_+xx] + fb[(yy+1)*MW_+xx+1]);
        }
        if (val > 0.5f) { bits |= (1u << (r + 1)); any = true; }
    }
    if (!any) bits |= 1u;
    allowed[s] = bits;
}

// ---------------------------------------------------------------------------
// Kernel 2: pack combined K (seg0=k, seg1/2=regional_k) and V into bf16
// WMMA B-fragment layout, so the attention kernel does pure b128 loads.
// B fragment (32K x 16N bf16, wave32): lane L: n = L%16, hi = L/16;
//   u32 j holds rows r = (j>=4?16:0) + (hi?8:0) + 2*(j&3) + {0,1}.
// K fragments: rows = head-dim chunk, cols = 16 keys.   frag(h,kt,c), c=dim chunk
// V fragments: rows = 32-key chunk,   cols = 16 dims.   frag(h,nt,kc)
// One thread produces one packed u32.
// ---------------------------------------------------------------------------
__global__ void pack_kv_kernel(const float* __restrict__ k, const float* __restrict__ rk,
                               const float* __restrict__ v, const float* __restrict__ rv,
                               unsigned* __restrict__ kfrag, unsigned* __restrict__ vfrag) {
    int tid = blockIdx.x * blockDim.x + threadIdx.x;
    const int KTOT = H_ * NKT * 2 * 256;     // 98304
    int e = (tid < KTOT) ? tid : tid - KTOT;
    int frag = e >> 8;
    int within = e & 255;
    int lane = within >> 3, j = within & 7;
    int n = lane & 15, hi = lane >> 4;
    int roff = ((j & 4) ? 16 : 0) + (hi ? 8 : 0) + 2 * (j & 3);

    if (tid < KTOT) {
        int c  = frag & 1;
        int kt = (frag >> 1) % NKT;
        int h  = frag / (2 * NKT);
        int key = kt * 16 + n;
        int seg = key >> 7, pp = key & 127;
        int d0 = c * 32 + roff;
        const float* src = (seg == 0) ? (k + ((size_t)(pp * H_ + h)) * D_)
                                      : (rk + ((size_t)(((seg - 1) * P_ + pp) * H_ + h)) * D_);
        kfrag[e] = pack_bf16_rne(src[d0], src[d0 + 1]);
    } else {
        int kc = frag % 12;
        int nt = (frag / 12) & 3;
        int h  = frag / 48;
        int d  = nt * 16 + n;
        int key0 = kc * 32 + roff;
        float a, b;
        {
            int seg = key0 >> 7, pp = key0 & 127;
            const float* src = (seg == 0) ? (v + ((size_t)(pp * H_ + h)) * D_)
                                          : (rv + ((size_t)(((seg - 1) * P_ + pp) * H_ + h)) * D_);
            a = src[d];
        }
        {
            int key1 = key0 + 1;
            int seg = key1 >> 7, pp = key1 & 127;
            const float* src = (seg == 0) ? (v + ((size_t)(pp * H_ + h)) * D_)
                                          : (rv + ((size_t)(((seg - 1) * P_ + pp) * H_ + h)) * D_);
            b = src[d];
        }
        vfrag[e] = pack_bf16_rne(a, b);
    }
}

// ---------------------------------------------------------------------------
// Kernel 3: fused regional+base attention.
// Block = 128 threads (4 wave32), each wave = 16 queries of one head.
//  1) S = (Q*scale) K^T for all 384 keys: 24 acc tiles in regs (48 bf16 WMMAs)
//  2) per-segment row maxima (base = segment 0's max, no separate pass);
//     masking folded into the exp subtrahend: mm = allowed ? m : +1e30
//     so exp(s-mm) underflows to 0 for disallowed entries (no per-elem selects)
//  3) combined P = 0.5*P_reg + 0.5*P_base(seg0) -> bf16 (truncate) LDS slab
//  4) O = P V over 384 keys (48 bf16 WMMAs), f32 out
// ---------------------------------------------------------------------------
__global__ void __launch_bounds__(128, 1)
attn_kernel(const float* __restrict__ q, const unsigned* __restrict__ allowed,
            const unsigned* __restrict__ kfrag, const unsigned* __restrict__ vfrag,
            float* __restrict__ out) {
    __shared__ unsigned short Pslab[WAVES][QROWS][NKEY];   // 48 KB

    const int lane = threadIdx.x & 31;
    const int wave = threadIdx.x >> 5;
    const int h    = blockIdx.x / (S_ / QBLK);
    const int qblk = blockIdx.x % (S_ / QBLK);
    const int qbase = qblk * QBLK + wave * QROWS;
    const int m  = lane & 15;       // A-fragment row / C-fragment col
    const int hi = lane >> 4;

    // ---- Q A-fragments (two 16x32 dim chunks), scale folded in (exact: 2^-3) ----
    Frag aq[2];
    #pragma unroll
    for (int c = 0; c < 2; ++c) {
        #pragma unroll
        for (int j = 0; j < 8; ++j) {
            int d = c * 32 + ((j & 4) ? 16 : 0) + (hi ? 8 : 0) + 2 * (j & 3);
            const float* qp = q + ((size_t)(qbase + m) * H_ + h) * D_ + d;
            aq[c].u[j] = pack_bf16_rne(qp[0] * SCALE, qp[1] * SCALE);
        }
    }

    // ---- scores: 24 register tiles of 16x16 (already scaled) ----
    v8f acc[NKT];
    #pragma unroll
    for (int kt = 0; kt < NKT; ++kt) {
        Frag b0 = load_frag(kfrag + (((size_t)(h * NKT + kt) * 2 + 0) << 8) + lane * 8);
        Frag b1 = load_frag(kfrag + (((size_t)(h * NKT + kt) * 2 + 1) << 8) + lane * 8);
        v8f c = {};
        c = wmma_bf16(aq[0].v, b0.v, c);
        c = wmma_bf16(aq[1].v, b1.v, c);
        acc[kt] = c;
    }

    // ---- per-row allowed-segment bits (C layout: u32 j -> row j + 8*hi) ----
    unsigned mb[8];
    #pragma unroll
    for (int j = 0; j < 8; ++j) mb[j] = allowed[qbase + j + hi * 8];

    // ---- per-segment local row maxima (pure fmax, no masking ops) ----
    float ms0[8], ms1[8], ms2[8];
    #pragma unroll
    for (int j = 0; j < 8; ++j) { ms0[j] = -1e30f; ms1[j] = -1e30f; ms2[j] = -1e30f; }
    #pragma unroll
    for (int kt = 0; kt < NKT; ++kt) {
        #pragma unroll
        for (int j = 0; j < 8; ++j) {
            float s = acc[kt][j];
            if      (kt < 8)  ms0[j] = fmaxf(ms0[j], s);
            else if (kt < 16) ms1[j] = fmaxf(ms1[j], s);
            else              ms2[j] = fmaxf(ms2[j], s);
        }
    }
    // allowed-segment selection BEFORE lane reduction (bits are row-uniform)
    float mreg[8], mbse[8];
    #pragma unroll
    for (int j = 0; j < 8; ++j) {
        float mr = -1e30f;
        mr = (mb[j] & 1u) ? fmaxf(mr, ms0[j]) : mr;
        mr = (mb[j] & 2u) ? fmaxf(mr, ms1[j]) : mr;
        mr = (mb[j] & 4u) ? fmaxf(mr, ms2[j]) : mr;
        float mbs = ms0[j];
        #pragma unroll
        for (int off = 1; off < 16; off <<= 1) {   // reduce within 16-lane half
            mr  = fmaxf(mr,  __shfl_xor(mr,  off, 32));
            mbs = fmaxf(mbs, __shfl_xor(mbs, off, 32));
        }
        mreg[j] = mr; mbse[j] = mbs;
    }
    // masked subtrahends: disallowed segment -> +1e30 so exp underflows to 0
    float mm0[8], mm1[8], mm2[8];
    #pragma unroll
    for (int j = 0; j < 8; ++j) {
        mm0[j] = (mb[j] & 1u) ? mreg[j] : 1e30f;
        mm1[j] = (mb[j] & 2u) ? mreg[j] : 1e30f;
        mm2[j] = (mb[j] & 4u) ? mreg[j] : 1e30f;
    }

    // ---- exponentials (cached) + row sums: pure sub+exp+add per element ----
    float lreg[8] = {0,0,0,0,0,0,0,0}, lbse[8] = {0,0,0,0,0,0,0,0};
    v8f ebase[8];
    #pragma unroll
    for (int kt = 0; kt < NKT; ++kt) {
        #pragma unroll
        for (int j = 0; j < 8; ++j) {
            float s = acc[kt][j];
            float mmv = (kt < 8) ? mm0[j] : ((kt < 16) ? mm1[j] : mm2[j]);
            float er = __expf(s - mmv);
            acc[kt][j] = er;             // cache regional exp (0 if disallowed)
            lreg[j] += er;
            if (kt < 8) {
                float eb = __expf(s - mbse[j]);
                ebase[kt][j] = eb;       // cache base exp
                lbse[j] += eb;
            }
        }
    }
    float rreg[8], rbse[8];
    #pragma unroll
    for (int j = 0; j < 8; ++j) {
        #pragma unroll
        for (int off = 1; off < 16; off <<= 1) {
            lreg[j] += __shfl_xor(lreg[j], off, 32);
            lbse[j] += __shfl_xor(lbse[j], off, 32);
        }
        rreg[j] = 0.5f / lreg[j];     // base_ratio folded in
        rbse[j] = 0.5f / lbse[j];
    }

    // ---- combined probabilities -> bf16 (truncate) LDS slab ----
    #pragma unroll
    for (int kt = 0; kt < NKT; ++kt) {
        #pragma unroll
        for (int j = 0; j < 8; ++j) {
            float p = acc[kt][j] * rreg[j];
            if (kt < 8) p += ebase[kt][j] * rbse[j];
            // truncating f32->bf16: store high 16 bits (ds_store_b16_d16_hi)
            Pslab[wave][j + hi * 8][kt * 16 + m] =
                (unsigned short)(__float_as_uint(p) >> 16);
        }
    }
    __syncthreads();

    // ---- O = P x V  (A from LDS slab, B fragments from packed global) ----
    v8f o[4] = {{}, {}, {}, {}};
    #pragma unroll
    for (int kc = 0; kc < 12; ++kc) {
        Frag ap;
        #pragma unroll
        for (int j = 0; j < 8; ++j) {
            int koff = kc * 32 + ((j & 4) ? 16 : 0) + (hi ? 8 : 0) + 2 * (j & 3);
            ap.u[j] = *(const unsigned*)&Pslab[wave][m][koff];
        }
        #pragma unroll
        for (int nt = 0; nt < 4; ++nt) {
            Frag bv = load_frag(vfrag + (((size_t)(h * 4 + nt) * 12 + kc) << 8) + lane * 8);
            o[nt] = wmma_bf16(ap.v, bv.v, o[nt]);
        }
    }

    // ---- write output [S, H*D] f32 (C layout) ----
    #pragma unroll
    for (int nt = 0; nt < 4; ++nt) {
        #pragma unroll
        for (int j = 0; j < 8; ++j) {
            out[(size_t)(qbase + j + hi * 8) * (H_ * D_) + h * D_ + nt * 16 + m] = o[nt][j];
        }
    }
}

// ---------------------------------------------------------------------------
// Launch
// ---------------------------------------------------------------------------
extern "C" void kernel_launch(void* const* d_in, const int* in_sizes, int n_in,
                              void* d_out, int out_size, void* d_ws, size_t ws_size,
                              hipStream_t stream) {
    const float* q  = (const float*)d_in[0];
    const float* k  = (const float*)d_in[1];
    const float* v  = (const float*)d_in[2];
    const float* rk = (const float*)d_in[3];
    const float* rv = (const float*)d_in[4];
    const float* rm = (const float*)d_in[5];
    float* out = (float*)d_out;

    unsigned* allowed = (unsigned*)d_ws;                 // 56320 u32
    unsigned* kfrag   = allowed + 57344;                 // padded; 98304 u32
    unsigned* vfrag   = kfrag + 98304;                   // 98304 u32

    mask_kernel<<<S_ / 256, 256, 0, stream>>>(rm, allowed);
    pack_kv_kernel<<<(2 * 98304) / 256, 256, 0, stream>>>(k, rk, v, rv, kfrag, vfrag);
    attn_kernel<<<H_ * (S_ / QBLK), 128, 0, stream>>>(q, allowed, kfrag, vfrag, out);
}
